// LlavaMultiModalModuleWrapper_33423435497652
// MI455X (gfx1250) — matrix operand order repaired and checked by reading, not compiled, not run on hardware
//
#include <hip/hip_runtime.h>

// ---------------- problem constants ----------------
#define IMAGE_TOKEN 32000
#define PAD_TOKEN   0
#define B_          8
#define S_          1024
#define IMG_        336
#define PATCH_      14
#define D_          4096
#define DV_         1024
#define GRID_       24          // IMG/PATCH
#define NP_         576         // 24*24 patches per image
#define MAXE_       1599        // NP-1+S
#define NIMG_       4608        // B*NP
#define K0RAW_      588         // 3*14*14
#define K0PAD_      608         // padded to multiple of 32
#define NLAYERS_    32
#define NEXPERTS_   8
#define NSHARED_    2

typedef unsigned short ushort_t;
typedef __attribute__((ext_vector_type(16))) __bf16        v16bf;
typedef __attribute__((ext_vector_type(8)))  float         v8f;
typedef __attribute__((ext_vector_type(4)))  unsigned int  u32x4;

union FragBF { v16bf v; u32x4 q[2]; };

__device__ __forceinline__ ushort_t f2bf(float f) {
    unsigned u = __float_as_uint(f);
    unsigned r = (u >> 16) & 1u;
    u += 0x7fffu + r;               // round-to-nearest-even
    return (ushort_t)(u >> 16);
}

__device__ __forceinline__ float gelu_tanh(float x) {
    const float k0 = 0.7978845608028654f;   // sqrt(2/pi)
    float inner = k0 * (x + 0.044715f * x * x * x);
    return 0.5f * x * (1.0f + tanhf(inner));
}

// Low 32 bits of a flat pointer into LDS == byte offset within the
// workgroup's LDS allocation (flat-LDS aperture keeps offset in addr[31:0]).
__device__ __forceinline__ unsigned lds_off(const void* p) {
    return (unsigned)(size_t)p;
}

// CDNA5 async copy: 16B per lane, global -> LDS, tracked by ASYNCcnt.
__device__ __forceinline__ void async_copy_b128(unsigned ldsAddr, const void* gaddr) {
    asm volatile("global_load_async_to_lds_b128 %0, %1, off"
                 :: "v"(ldsAddr), "v"(gaddr) : "memory");
}
__device__ __forceinline__ void wait_asynccnt0() {
    asm volatile("s_wait_asynccnt 0x0" ::: "memory");
}

// ---------------- init kernels ----------------
__global__ void init_out_kernel(float* out, long long total,
                                long long labOff, long long labN) {
    long long i = (long long)blockIdx.x * blockDim.x + threadIdx.x;
    long long stride = (long long)gridDim.x * blockDim.x;
    for (; i < total; i += stride)
        out[i] = (i >= labOff && i < labOff + labN) ? -100.0f : 0.0f;
}

__global__ void init_meta_kernel(int* written, int* img_map) {
    int i = blockIdx.x * blockDim.x + threadIdx.x;
    if (i < B_ * MAXE_) written[i] = 0;
    if (i < NIMG_)      img_map[i] = -1;
}

__global__ void leftpad_kernel(const int* ids, int* flag) {
    if (threadIdx.x == 0 && blockIdx.x == 0) {
        int c = 0;
        for (int b = 0; b < B_; ++b) c += (ids[b * S_ + S_ - 1] == PAD_TOKEN);
        *flag = (c == 0) ? 1 : 0;
    }
}

// ---------------- merge position scan (one block per batch row) ----------------
__global__ __launch_bounds__(1024)
void pos_scan_kernel(const int* ids, int* pos_safe, int* written,
                     int* nbpad, const int* leftpad) {
    __shared__ int sc[S_];
    __shared__ int pad_s;
    int b = blockIdx.x;
    int s = threadIdx.x;
    int tok = ids[b * S_ + s];
    int special = (tok == IMAGE_TOKEN) ? 1 : 0;
    sc[s] = special ? NP_ : 1;      // special*(p-1)+1
    __syncthreads();
    for (int off = 1; off < S_; off <<= 1) {
        int add = (s >= off) ? sc[s - off] : 0;
        __syncthreads();
        sc[s] += add;
        __syncthreads();
    }
    int newpos = sc[s] - 1;
    if (s == S_ - 1) {
        int nb = (MAXE_ - 1) - newpos;
        nbpad[b] = nb;
        pad_s = (*leftpad) ? nb : 0;
    }
    __syncthreads();
    int p = newpos + pad_s;
    int ps = special ? MAXE_ : p;
    pos_safe[b * S_ + s] = ps;
    if (ps >= 0 && ps < MAXE_) written[b * MAXE_ + ps] = 1;
}

// ---------------- image slot ranking (tiny, single thread) ----------------
__global__ void img_slots_kernel(const int* written, const int* nbpad,
                                 int* img_map, float* attnOut) {
    if (threadIdx.x != 0 || blockIdx.x != 0) return;
    int rank = 0;
    for (int b = 0; b < B_; ++b) {
        int cz = 0;
        int nb = nbpad[b];
        for (int e = 0; e < MAXE_; ++e) {
            if (!written[b * MAXE_ + e]) {
                if (cz >= nb) {
                    if (rank < NIMG_) img_map[rank] = b * MAXE_ + e;
                    attnOut[b * MAXE_ + e] = 1.0f;
                    ++rank;
                }
                ++cz;
            }
        }
    }
}

// ---------------- text embedding scatter ----------------
__global__ __launch_bounds__(256)
void scatter_text_kernel(const int* ids, const int* am, const int* lab,
                         const float* table, const int* pos_safe,
                         float* embOut, float* attnOut, float* labOut) {
    int idx = blockIdx.x;               // b*S + s
    int b = idx / S_;
    int ps = pos_safe[idx];
    if (ps < 0 || ps >= MAXE_) return;
    int tok = ids[idx];
    const float* src = table + (size_t)tok * D_;
    float* dst = embOut + ((size_t)b * MAXE_ + ps) * D_;
    for (int j = threadIdx.x; j < D_; j += blockDim.x) dst[j] = src[j];
    if (threadIdx.x == 0) {
        attnOut[b * MAXE_ + ps] = (float)am[idx];
        labOut[b * MAXE_ + ps]  = (float)lab[idx];
    }
}

// ---------------- bf16 conversion with transpose ----------------
// src: f32 [K][N] row-major; dst: bf16 [N][K] (so GEMM B-tiles are contiguous)
__global__ void cvt_bf16_t_kernel(const float* s, ushort_t* d, int N, int K) {
    long long total = (long long)N * K;
    long long i = (long long)blockIdx.x * blockDim.x + threadIdx.x;
    long long stride = (long long)gridDim.x * blockDim.x;
    for (; i < total; i += stride) {
        int n = (int)(i / K);
        int k = (int)(i % K);
        d[i] = f2bf(s[(size_t)k * N + n]);
    }
}

// patch_w: f32 [588][1024] -> bf16 [1024][608] transposed + K zero-padded
__global__ void pack_patchw_t_kernel(const float* s, ushort_t* d) {
    int i = blockIdx.x * blockDim.x + threadIdx.x;      // over DV_*K0PAD_
    if (i >= DV_ * K0PAD_) return;
    int n = i / K0PAD_, k = i % K0PAD_;
    d[i] = (k < K0RAW_) ? f2bf(s[(size_t)k * DV_ + n]) : (ushort_t)0;
}

__global__ __launch_bounds__(256)
void patchify_kernel(const float* px, ushort_t* Xbf) {
    int row = blockIdx.x;               // 0..NIMG_-1
    int n  = row / NP_;
    int pr = row % NP_;
    int gy = pr / GRID_, gx = pr % GRID_;
    for (int k = threadIdx.x; k < K0PAD_; k += blockDim.x) {
        float v = 0.f;
        if (k < K0RAW_) {
            int c = k / (PATCH_ * PATCH_);
            int rem = k % (PATCH_ * PATCH_);
            int py = rem / PATCH_, pxx = rem % PATCH_;
            v = px[(((size_t)n * 3 + c) * IMG_ + (gy * PATCH_ + py)) * IMG_
                   + gx * PATCH_ + pxx];
        }
        Xbf[(size_t)row * K0PAD_ + k] = f2bf(v);
    }
}

// ---------------- WMMA bf16 GEMM: C = A(MxK) * Bt(NxK)^T + bias ----------------
// A row-major [M][K]; Bt pre-transposed [N][K]. Block tile 128x64, 8 wave32
// waves, each wave a 32x32 tile (2x2 WMMA). Double-buffered LDS filled with
// GLOBAL_LOAD_ASYNC_TO_LDS_B128 (ASYNCcnt), one barrier per K-step.
#define BM 128
#define BN 64
#define BK 32

template<bool GELU, bool SCATTER>
__global__ __launch_bounds__(256)
void gemm_bf16_kernel(const ushort_t* __restrict__ A,
                      const ushort_t* __restrict__ Bt,
                      const float* __restrict__ bias,
                      ushort_t* __restrict__ outBf,
                      float* __restrict__ outF,
                      const int* __restrict__ rowmap,
                      int M, int N, int K) {
    __shared__ ushort_t Asm[2][BM][BK];    // 2 x 8 KB
    __shared__ ushort_t Bsm[2][BN][BK];    // 2 x 4 KB   [n][k]
    const int bm = blockIdx.y * BM;
    const int bn = blockIdx.x * BN;
    const int t = threadIdx.x;
    const int wave = t >> 5;
    const int lane = t & 31;
    const int wr = wave >> 1;           // 0..3
    const int wc = wave & 1;            // 0..1
    const int halfsel = lane >> 4;
    const int lrow = lane & 15;

    // A tile: thread t copies 32 B (two b128) of row t>>1
    const int arow_ld = t >> 1;
    const int acol_ld = (t & 1) * 16;
    // B tile: thread t copies 16 B of row t>>2 of Bt
    const int brow_ld = t >> 2;
    const int bcol_ld = (t & 3) * 8;

    auto load_stage = [&](int k0, int buf) {
        const ushort_t* asrc = A + (size_t)(bm + arow_ld) * K + k0 + acol_ld;
        async_copy_b128(lds_off(&Asm[buf][arow_ld][acol_ld]),     asrc);
        async_copy_b128(lds_off(&Asm[buf][arow_ld][acol_ld + 8]), asrc + 8);
        const ushort_t* bsrc = Bt + (size_t)(bn + brow_ld) * K + k0 + bcol_ld;
        async_copy_b128(lds_off(&Bsm[buf][brow_ld][bcol_ld]),     bsrc);
    };

    v8f acc[2][2] = {};
    const int nk = K / BK;

    load_stage(0, 0);
    for (int s = 0; s < nk; ++s) {
        const int buf = s & 1;
        wait_asynccnt0();               // own stage-s copies landed
        __syncthreads();                // everyone's stage-s data visible
        if (s + 1 < nk) load_stage((s + 1) * BK, buf ^ 1);

        #pragma unroll
        for (int mt = 0; mt < 2; ++mt) {
            FragBF a;
            int ar = wr * 32 + mt * 16 + lrow;
            a.q[0] = *(const u32x4*)&Asm[buf][ar][halfsel * 8];       // K 0-7 / 8-15
            a.q[1] = *(const u32x4*)&Asm[buf][ar][16 + halfsel * 8];  // K 16-23 / 24-31
            #pragma unroll
            for (int nt = 0; nt < 2; ++nt) {
                FragBF b;
                int bc = wc * 32 + nt * 16 + lrow;
                b.q[0] = *(const u32x4*)&Bsm[buf][bc][halfsel * 8];       // K 0-7 / 8-15
                b.q[1] = *(const u32x4*)&Bsm[buf][bc][16 + halfsel * 8];  // K 16-23 / 24-31
                acc[mt][nt] = __builtin_amdgcn_wmma_f32_16x16x32_bf16(
                    false, a.v, false, b.v, (short)0, acc[mt][nt], false, false);
            }
        }
    }

    // epilogue: C layout -> row = i + halfsel*8, col = lane&15 per 16x16 tile
    #pragma unroll
    for (int mt = 0; mt < 2; ++mt) {
        #pragma unroll
        for (int nt = 0; nt < 2; ++nt) {
            int colg = bn + wc * 32 + nt * 16 + lrow;
            float bv = bias[colg];
            #pragma unroll
            for (int i = 0; i < 8; ++i) {
                int rowg = bm + wr * 32 + mt * 16 + halfsel * 8 + i;
                float v = acc[mt][nt][i] + bv;
                if (GELU) v = gelu_tanh(v);
                if (!SCATTER) {
                    outBf[(size_t)rowg * N + colg] = f2bf(v);
                } else {
                    int dst = rowmap[rowg];
                    if (dst >= 0)
                        outF[(size_t)dst * (size_t)N + colg] = v;
                }
            }
        }
    }
}

// ---------------- launcher ----------------
extern "C" void kernel_launch(void* const* d_in, const int* in_sizes, int n_in,
                              void* d_out, int out_size, void* d_ws, size_t ws_size,
                              hipStream_t stream) {
    const int*   ids     = (const int*)d_in[0];
    const float* px      = (const float*)d_in[1];
    const int*   am      = (const int*)d_in[2];
    const int*   lab     = (const int*)d_in[3];
    const float* table   = (const float*)d_in[4];
    // d_in[5] = cls_emb: unused (reference drops the CLS token before proj)
    const float* patch_w = (const float*)d_in[6];
    const float* patch_b = (const float*)d_in[7];
    const float* w1      = (const float*)d_in[8];
    const float* b1      = (const float*)d_in[9];
    const float* w2      = (const float*)d_in[10];
    const float* b2      = (const float*)d_in[11];

    float* out = (float*)d_out;
    const long long EMB_OFF  = 1;
    const long long EMB_N    = (long long)B_ * MAXE_ * D_;
    const long long ATTN_OFF = EMB_OFF + EMB_N;
    const long long ATTN_N   = (long long)B_ * MAXE_;
    const long long LAB_OFF  = ATTN_OFF + ATTN_N;
    const long long LAB_N    = (long long)B_ * MAXE_;
    const long long TOTAL    = LAB_OFF + LAB_N
                             + (long long)NLAYERS_ * MAXE_ * NEXPERTS_
                             + (long long)NLAYERS_ * MAXE_ * NSHARED_;

    char* ws = (char*)d_ws;
    size_t off = 0;
    auto alloc = [&](size_t bytes) -> char* {
        char* p = ws + off;
        off = (off + bytes + 255) & ~(size_t)255;
        return p;
    };
    ushort_t* Xbf  = (ushort_t*)alloc((size_t)NIMG_ * K0PAD_ * 2);
    ushort_t* PWbf = (ushort_t*)alloc((size_t)DV_ * K0PAD_ * 2);   // [1024][608]
    ushort_t* W1bf = (ushort_t*)alloc((size_t)D_ * DV_ * 2);       // [4096][1024]
    ushort_t* W2bf = (ushort_t*)alloc((size_t)D_ * D_ * 2);        // [4096][4096]
    ushort_t* Hbf  = (ushort_t*)alloc((size_t)NIMG_ * DV_ * 2);
    ushort_t* Gbf  = (ushort_t*)alloc((size_t)NIMG_ * D_ * 2);
    int* pos_safe  = (int*)alloc((size_t)B_ * S_ * 4);
    int* written   = (int*)alloc((size_t)B_ * MAXE_ * 4);
    int* nbpad     = (int*)alloc((size_t)B_ * 4);
    int* lp        = (int*)alloc(4);
    int* imap      = (int*)alloc((size_t)NIMG_ * 4);
    if (off > ws_size) return;   // insufficient scratch

    // 1. init output (zeros; labels region = -100)
    init_out_kernel<<<8192, 256, 0, stream>>>(out, TOTAL, LAB_OFF, LAB_N);
    // 2. init metadata
    init_meta_kernel<<<(B_ * MAXE_ + 255) / 256, 256, 0, stream>>>(written, imap);
    // 3. left padding flag
    leftpad_kernel<<<1, 32, 0, stream>>>(ids, lp);
    // 4. token position scan
    pos_scan_kernel<<<B_, 1024, 0, stream>>>(ids, pos_safe, written, nbpad, lp);
    // 5. rank image slots + set their attention
    img_slots_kernel<<<1, 32, 0, stream>>>(written, nbpad, imap, out + ATTN_OFF);
    // 6. scatter text embeddings / attn / labels
    scatter_text_kernel<<<B_ * S_, 256, 0, stream>>>(
        ids, am, lab, table, pos_safe,
        out + EMB_OFF, out + ATTN_OFF, out + LAB_OFF);
    // 7-9. convert weights to bf16, pre-transposed to [N][K]
    cvt_bf16_t_kernel<<<4096, 256, 0, stream>>>(w1, W1bf, D_, DV_);
    cvt_bf16_t_kernel<<<8192, 256, 0, stream>>>(w2, W2bf, D_, D_);
    pack_patchw_t_kernel<<<(DV_ * K0PAD_ + 255) / 256, 256, 0, stream>>>(patch_w, PWbf);
    // 10. patchify pixels -> bf16 activations (K padded with zeros)
    patchify_kernel<<<NIMG_, 256, 0, stream>>>(px, Xbf);
    // 11. H = X @ patch_w + patch_b            (4608x608 @ 608x1024)
    gemm_bf16_kernel<false, false><<<dim3(DV_ / BN, NIMG_ / BM), 256, 0, stream>>>(
        Xbf, PWbf, patch_b, Hbf, nullptr, nullptr, NIMG_, DV_, K0PAD_);
    // 12. G = gelu(H @ w1 + b1)                (4608x1024 @ 1024x4096)
    gemm_bf16_kernel<true, false><<<dim3(D_ / BN, NIMG_ / BM), 256, 0, stream>>>(
        Hbf, W1bf, b1, Gbf, nullptr, nullptr, NIMG_, D_, DV_);
    // 13. F = G @ w2 + b2, scattered into final_emb rows  (4608x4096 @ 4096x4096)
    gemm_bf16_kernel<false, true><<<dim3(D_ / BN, NIMG_ / BM), 256, 0, stream>>>(
        Gbf, W2bf, b2, nullptr, out + EMB_OFF, imap, NIMG_, D_, D_);
}
